// DecoderRNN_73976516706718
// MI455X (gfx1250) — matrix operand order repaired and checked by reading, not compiled, not run on hardware
//
#include <hip/hip_runtime.h>

typedef __attribute__((ext_vector_type(16))) __bf16 v16bf;
typedef __attribute__((ext_vector_type(8)))  __bf16 v8bf;
typedef __attribute__((ext_vector_type(8)))  float  v8f;

#define B_   128
#define N_   49
#define E_   2048
#define EMB_ 512
#define D_   512
#define V_   12000
#define T_   21
#define XH_  3072   // context(2048) | embed(512) | h(512)

// ---------------------------------------------------------------------------
// Generic bf16 GEMM:  C[M,N] = A[M,K] @ W[N,K]^T + bias[N]
// A row-major (lda=K), W row-major (ldw=K), C row-major with ldc.
// Wave computes a 16x64 tile via 4x v_wmma_f32_16x16x32_bf16 per K-step.
// Block = 128 threads = 4 waves stacked in M -> 64x64 per block.
// ---------------------------------------------------------------------------
__launch_bounds__(128)
__global__ void gemm_bf16_xwt(const __bf16* __restrict__ A,
                              const __bf16* __restrict__ W,
                              const float*  __restrict__ bias,
                              float*        __restrict__ C,
                              int M, int N, int K, int ldc)
{
    const int lane = threadIdx.x & 31;
    const int wave = threadIdx.x >> 5;
    const int m0 = (blockIdx.x * 4 + wave) * 16;
    const int n0 = blockIdx.y * 64;
    if (m0 >= M) return;

    const int l15 = lane & 15;
    const int h   = lane >> 4;          // K-half select (this code's k = 16h + j order)
    const int mrow = m0 + l15;

    v8f acc[4];
    for (int t = 0; t < 4; ++t)
        for (int j = 0; j < 8; ++j) acc[t][j] = 0.0f;

    int ncol[4];
    for (int t = 0; t < 4; ++t) {
        int n = n0 + t * 16 + l15;
        ncol[t] = (n < N) ? n : (N - 1);   // clamp load (store is guarded)
    }

    const __bf16* arow = A + (size_t)mrow * K;
    for (int k0 = 0; k0 < K; k0 += 32) {
        // A fragment: lane holds row m, elem j -> K=k0+h*16+j
        v8bf alo = *reinterpret_cast<const v8bf*>(arow + k0 + h * 16);
        v8bf ahi = *reinterpret_cast<const v8bf*>(arow + k0 + h * 16 + 8);
        v16bf a;
        for (int j = 0; j < 8; ++j) { a[j] = alo[j]; a[8 + j] = ahi[j]; }

        for (int t = 0; t < 4; ++t) {
            // B fragment: lane holds column n (= W row n), elem j -> K=k0+h*16+j
            const __bf16* wp = W + (size_t)ncol[t] * K + k0 + h * 16;
            v16bf b = *reinterpret_cast<const v16bf*>(wp);
            acc[t] = __builtin_amdgcn_wmma_f32_16x16x32_bf16(
                false, a, false, b, (short)0, acc[t], false, false);
        }
    }

    // D layout: lane l, vgpr r -> M = m0 + (l>>4)*8 + r, N = n0 + t*16 + (l&15)
    for (int t = 0; t < 4; ++t) {
        int n = n0 + t * 16 + l15;
        if (n >= N) continue;
        float bv = bias ? bias[n] : 0.0f;
        for (int r = 0; r < 8; ++r) {
            int row = m0 + h * 8 + r;
            C[(size_t)row * ldc + n] = acc[t][r] + bv;
        }
    }
}

// ---------------------------------------------------------------------------
// Helper kernels
// ---------------------------------------------------------------------------
__global__ void f2b_kernel(const float* __restrict__ in, __bf16* __restrict__ out, size_t n)
{
    size_t i = (size_t)blockIdx.x * blockDim.x + threadIdx.x;
    size_t stride = (size_t)gridDim.x * blockDim.x;
    for (; i < n; i += stride) out[i] = (__bf16)in[i];
}

__global__ void pack_wcat_kernel(const float* __restrict__ W_ih,
                                 const float* __restrict__ W_hh,
                                 __bf16* __restrict__ wcat)
{
    size_t n = (size_t)2048 * XH_;
    size_t i = (size_t)blockIdx.x * blockDim.x + threadIdx.x;
    size_t stride = (size_t)gridDim.x * blockDim.x;
    for (; i < n; i += stride) {
        size_t r = i / XH_, c = i % XH_;
        float v = (c < 2560) ? W_ih[r * 2560 + c] : W_hh[r * 512 + (c - 2560)];
        wcat[i] = (__bf16)v;
    }
}

__global__ void bias_cat_kernel(const float* __restrict__ b_ih,
                                const float* __restrict__ b_hh,
                                float* __restrict__ bcat)
{
    int i = blockIdx.x * blockDim.x + threadIdx.x;
    if (i < 2048) bcat[i] = b_ih[i] + b_hh[i];
}

__global__ void gather_embed_kernel(const int* __restrict__ captions,
                                    const float* __restrict__ embedding,
                                    __bf16* __restrict__ embeds)
{
    int idx = blockIdx.x * blockDim.x + threadIdx.x;   // B_*T_*EMB_
    int j  = idx & (EMB_ - 1);
    int bt = idx >> 9;                                  // b*T_ + t
    int b = bt / T_, t = bt - b * T_;
    int tok = captions[b * (T_ + 1) + t];               // captions[:, :-1]
    embeds[(size_t)bt * EMB_ + j] = (__bf16)embedding[(size_t)tok * EMB_ + j];
}

__global__ void mean_feat_kernel(const float* __restrict__ feats,
                                 __bf16* __restrict__ meanf)
{
    int b = blockIdx.x;
    int e = blockIdx.y * blockDim.x + threadIdx.x;
    const float* fb = feats + (size_t)b * N_ * E_ + e;
    float s = 0.f;
    for (int n = 0; n < N_; ++n) s += fb[(size_t)n * E_];
    meanf[(size_t)b * E_ + e] = (__bf16)(s * (1.0f / (float)N_));
}

// attention: scores[b,n] = Va . tanh(f_proj[b,n,:] + h_proj[b,:]) + Va_b ; softmax over n
__launch_bounds__(256)
__global__ void attn_kernel(const float* __restrict__ f_proj,
                            const float* __restrict__ h_proj,
                            const float* __restrict__ va_w,
                            const float* __restrict__ va_b,
                            float* __restrict__ weights, int t)
{
    __shared__ float sc[N_];
    __shared__ float red[2];
    int b = blockIdx.x;
    int lane = threadIdx.x & 31;
    int wave = threadIdx.x >> 5;
    const float* hp = h_proj + (size_t)b * D_;

    for (int n = wave; n < N_; n += 8) {
        const float* fp = f_proj + ((size_t)b * N_ + n) * D_;
        float acc = 0.f;
        for (int d = lane; d < D_; d += 32)
            acc += va_w[d] * tanhf(fp[d] + hp[d]);
        for (int off = 16; off > 0; off >>= 1)
            acc += __shfl_xor(acc, off);
        if (lane == 0) sc[n] = acc + va_b[0];
    }
    __syncthreads();
    if (threadIdx.x == 0) {
        float m = sc[0];
        for (int n = 1; n < N_; ++n) m = fmaxf(m, sc[n]);
        float s = 0.f;
        for (int n = 0; n < N_; ++n) s += expf(sc[n] - m);
        red[0] = m; red[1] = 1.0f / s;
    }
    __syncthreads();
    if (threadIdx.x < N_) {
        int n = threadIdx.x;
        weights[((size_t)b * T_ + t) * N_ + n] = expf(sc[n] - red[0]) * red[1];
    }
}

// context[b,e] = sum_n w[b,n]*features[b,n,e]  -> xh[:, 0:2048] (bf16)
__global__ void context_kernel(const float* __restrict__ feats,
                               const float* __restrict__ weights,
                               __bf16* __restrict__ xh, int t)
{
    int b = blockIdx.x;
    int e = blockIdx.y * blockDim.x + threadIdx.x;
    const float* wr = weights + ((size_t)b * T_ + t) * N_;
    const float* fb = feats + (size_t)b * N_ * E_ + e;
    float s = 0.f;
    for (int n = 0; n < N_; ++n) s += wr[n] * fb[(size_t)n * E_];
    xh[(size_t)b * XH_ + e] = (__bf16)s;
}

// xh[:, 2048:2560] = embed_t ; xh[:, 2560:3072] = h
__global__ void pack_xh_kernel(const __bf16* __restrict__ embeds,
                               const __bf16* __restrict__ hbf,
                               __bf16* __restrict__ xh, int t)
{
    int idx = blockIdx.x * blockDim.x + threadIdx.x;   // B_*1024
    int b = idx >> 10;
    int j = idx & 1023;
    if (j < 512)
        xh[(size_t)b * XH_ + 2048 + j] = embeds[((size_t)b * T_ + t) * EMB_ + j];
    else
        xh[(size_t)b * XH_ + 2560 + (j - 512)] = hbf[(size_t)b * D_ + (j - 512)];
}

__global__ void lstm_kernel(const float* __restrict__ gates,
                            float* __restrict__ c_state,
                            __bf16* __restrict__ hbf)
{
    int idx = blockIdx.x * blockDim.x + threadIdx.x;   // B_*D_
    int b = idx >> 9;
    int d = idx & 511;
    const float* g = gates + (size_t)b * 2048;
    float i  = 1.f / (1.f + expf(-g[d]));
    float f  = 1.f / (1.f + expf(-g[512 + d]));
    float gg = tanhf(g[1024 + d]);
    float o  = 1.f / (1.f + expf(-g[1536 + d]));
    float c  = f * c_state[idx] + i * gg;
    c_state[idx] = c;
    hbf[idx] = (__bf16)(o * tanhf(c));
}

// ---------------------------------------------------------------------------
extern "C" void kernel_launch(void* const* d_in, const int* in_sizes, int n_in,
                              void* d_out, int out_size, void* d_ws, size_t ws_size,
                              hipStream_t stream)
{
    const float* features  = (const float*)d_in[0];
    const int*   captions  = (const int*)  d_in[1];
    const float* embedding = (const float*)d_in[2];
    const float* Wa_w = (const float*)d_in[3];
    const float* Wa_b = (const float*)d_in[4];
    const float* Ua_w = (const float*)d_in[5];
    const float* Ua_b = (const float*)d_in[6];
    const float* Va_w = (const float*)d_in[7];
    const float* Va_b = (const float*)d_in[8];
    const float* W_ih = (const float*)d_in[9];
    const float* W_hh = (const float*)d_in[10];
    const float* b_ih = (const float*)d_in[11];
    const float* b_hh = (const float*)d_in[12];
    const float* ih_w = (const float*)d_in[13];
    const float* ih_b = (const float*)d_in[14];
    const float* ic_w = (const float*)d_in[15];
    const float* ic_b = (const float*)d_in[16];
    const float* fc_w = (const float*)d_in[17];
    const float* fc_b = (const float*)d_in[18];

    float* out_f       = (float*)d_out;
    float* weights_out = out_f;                                  // (B,T,N)
    float* preds_out   = out_f + (size_t)B_ * T_ * N_;           // (B,T,V)

    // ---- workspace carve-up (256B aligned) ----
    char* ws = (char*)d_ws;
    size_t off = 0;
    auto carve = [&](size_t bytes) -> void* {
        void* p = ws + off;
        off += (bytes + 255) & ~(size_t)255;
        return p;
    };
    __bf16* feat_bf  = (__bf16*)carve((size_t)B_ * N_ * E_ * 2);
    __bf16* wa_bf    = (__bf16*)carve((size_t)D_ * E_ * 2);
    __bf16* ua_bf    = (__bf16*)carve((size_t)D_ * D_ * 2);
    __bf16* ihw_bf   = (__bf16*)carve((size_t)D_ * E_ * 2);
    __bf16* icw_bf   = (__bf16*)carve((size_t)D_ * E_ * 2);
    __bf16* fcw_bf   = (__bf16*)carve((size_t)V_ * D_ * 2);
    __bf16* wcat_bf  = (__bf16*)carve((size_t)2048 * XH_ * 2);
    float*  bcat_f   = (float*) carve((size_t)2048 * 4);
    __bf16* emb_bf   = (__bf16*)carve((size_t)B_ * T_ * EMB_ * 2);
    __bf16* meanf_bf = (__bf16*)carve((size_t)B_ * E_ * 2);
    float*  fproj_f  = (float*) carve((size_t)B_ * N_ * D_ * 4);
    float*  h0_f     = (float*) carve((size_t)B_ * D_ * 4);
    __bf16* h_bf     = (__bf16*)carve((size_t)B_ * D_ * 2);
    float*  c_f      = (float*) carve((size_t)B_ * D_ * 4);
    float*  hproj_f  = (float*) carve((size_t)B_ * D_ * 4);
    __bf16* xh_bf    = (__bf16*)carve((size_t)B_ * XH_ * 2);
    float*  gates_f  = (float*) carve((size_t)B_ * 2048 * 4);
    (void)ws_size; (void)in_sizes; (void)n_in; (void)out_size;

    // ---- one-time conversions / packing ----
    f2b_kernel<<<2048, 256, 0, stream>>>(features, feat_bf, (size_t)B_ * N_ * E_);
    f2b_kernel<<<1024, 256, 0, stream>>>(Wa_w, wa_bf, (size_t)D_ * E_);
    f2b_kernel<<<512,  256, 0, stream>>>(Ua_w, ua_bf, (size_t)D_ * D_);
    f2b_kernel<<<1024, 256, 0, stream>>>(ih_w, ihw_bf, (size_t)D_ * E_);
    f2b_kernel<<<1024, 256, 0, stream>>>(ic_w, icw_bf, (size_t)D_ * E_);
    f2b_kernel<<<2048, 256, 0, stream>>>(fc_w, fcw_bf, (size_t)V_ * D_);
    pack_wcat_kernel<<<2048, 256, 0, stream>>>(W_ih, W_hh, wcat_bf);
    bias_cat_kernel<<<8, 256, 0, stream>>>(b_ih, b_hh, bcat_f);
    gather_embed_kernel<<<(B_ * T_ * EMB_) / 256, 256, 0, stream>>>(captions, embedding, emb_bf);
    mean_feat_kernel<<<dim3(B_, E_ / 256), 256, 0, stream>>>(features, meanf_bf);

    dim3 gblk(128);
    auto gemm = [&](const __bf16* A, const __bf16* W, const float* bias, float* C,
                    int M, int N, int K, int ldc) {
        dim3 grid((M + 63) / 64, (N + 63) / 64);
        gemm_bf16_xwt<<<grid, gblk, 0, stream>>>(A, W, bias, C, M, N, K, ldc);
    };

    // f_proj = features @ Wa_w^T + Wa_b   (6272 x 512, K=2048)
    gemm(feat_bf, wa_bf, Wa_b, fproj_f, B_ * N_, D_, E_, D_);
    // h0 / c0
    gemm(meanf_bf, ihw_bf, ih_b, h0_f, B_, D_, E_, D_);
    gemm(meanf_bf, icw_bf, ic_b, c_f,  B_, D_, E_, D_);
    f2b_kernel<<<256, 256, 0, stream>>>(h0_f, h_bf, (size_t)B_ * D_);

    // ---- recurrence ----
    for (int t = 0; t < T_; ++t) {
        // h_proj = h @ Ua^T + Ua_b
        gemm(h_bf, ua_bf, Ua_b, hproj_f, B_, D_, D_, D_);
        // attention scores + softmax -> weights output slice
        attn_kernel<<<B_, 256, 0, stream>>>(fproj_f, hproj_f, Va_w, Va_b, weights_out, t);
        // context -> xh[:, 0:2048]
        context_kernel<<<dim3(B_, E_ / 256), 256, 0, stream>>>(features, weights_out, xh_bf, t);
        // embed_t, h -> xh[:, 2048:3072]
        pack_xh_kernel<<<(B_ * 1024) / 256, 256, 0, stream>>>(emb_bf, h_bf, xh_bf, t);
        // gates = xh @ [W_ih | W_hh]^T + (b_ih + b_hh)
        gemm(xh_bf, wcat_bf, bcat_f, gates_f, B_, 2048, XH_, 2048);
        // LSTM pointwise -> updates c (f32), h (bf16)
        lstm_kernel<<<(B_ * D_) / 256, 256, 0, stream>>>(gates_f, c_f, h_bf);
        // preds[:, t, :] = h2 @ fc_w^T + fc_b
        gemm(h_bf, fcw_bf, fc_b, preds_out + (size_t)t * V_, B_, V_, D_, T_ * V_);
    }
}
